// FPN_AAR_DYN_25623774888402
// MI455X (gfx1250) — matrix operand order repaired and checked by reading, not compiled, not run on hardware
//
#include <hip/hip_runtime.h>
#include <hip/hip_bf16.h>
#include <stdint.h>

// ---------------------------------------------------------------------------
// Types for CDNA5 WMMA (wave32): V_WMMA_F32_16X16X32_BF16
// ---------------------------------------------------------------------------
typedef __attribute__((ext_vector_type(16))) __bf16 v16bf;
typedef __attribute__((ext_vector_type(8)))  __bf16 v8bf;
typedef __attribute__((ext_vector_type(8)))  float  v8f;

#define BS     8
#define CIN    256
#define COUT   256
#define IMH    128
#define IMW    128
#define HID    16
#define NK     5

// workspace layout (float element offsets)
#define OFF_POOLED 0        // 8*256
#define OFF_NATT   2176     // 8*5
#define OFF_CIN    2304     // 8*256
#define OFF_K2     4352     // 8*9
#define OFF_OUTA   4608     // 8*256
#define OFF_AGGB   6656     // 8*256
#define OFF_WBF_BYTES 34816 // bf16 agg weights: [b][rs][o][c], 8*9*256*256 ushorts

__device__ __forceinline__ unsigned short f32_to_bf16_rne(float f) {
    unsigned int u = __float_as_uint(f);
    u += 0x7FFFu + ((u >> 16) & 1u);
    return (unsigned short)(u >> 16);
}

#define SHUF16(lo, hi) __builtin_shufflevector(lo, hi, 0,1,2,3,4,5,6,7,8,9,10,11,12,13,14,15)

// Rotation source maps: flat spatial index (r*3+s) of OLD tap feeding NEW tap,
// for branches k = 0 (identity), 45, 90, 135, 180 degrees.
__constant__ int ROT[NK][9] = {
    {0,1,2, 3,4,5, 6,7,8},
    {3,0,1, 6,4,2, 7,8,5},
    {6,3,0, 7,4,1, 8,5,2},
    {7,6,3, 8,4,0, 5,2,1},
    {8,7,6, 5,4,3, 2,1,0}};

// ---------------------------------------------------------------------------
// 1) global average pool: one block per (b,c)
// ---------------------------------------------------------------------------
__global__ __launch_bounds__(256) void k_pool(const float* __restrict__ x,
                                              float* __restrict__ ws) {
    __shared__ float red[256];
    int bc = blockIdx.x;
    const float* p = x + (size_t)bc * (IMH * IMW);
    float s = 0.f;
    for (int i = threadIdx.x; i < IMH * IMW; i += 256) s += p[i];
    red[threadIdx.x] = s;
    __syncthreads();
    for (int off = 128; off > 0; off >>= 1) {
        if (threadIdx.x < (unsigned)off) red[threadIdx.x] += red[threadIdx.x + off];
        __syncthreads();
    }
    if (threadIdx.x == 0) ws[OFF_POOLED + bc] = red[0] * (1.0f / (IMH * IMW));
}

// ---------------------------------------------------------------------------
// 2) attention head + aggregated bias: one block per sample
// ---------------------------------------------------------------------------
__global__ __launch_bounds__(256) void k_attn(const float* __restrict__ net_w,
                                              const float* __restrict__ n_fc_w,
                                              const float* __restrict__ cin_fc_w,
                                              const float* __restrict__ k2_fc_w,
                                              const float* __restrict__ out_fc_w,
                                              const float* __restrict__ base_b,
                                              const float* __restrict__ extra_b,
                                              float* __restrict__ ws) {
    __shared__ float sp[256];
    __shared__ float sh[HID];
    __shared__ float slog[NK];
    int b = blockIdx.x, t = threadIdx.x;
    sp[t] = ws[OFF_POOLED + b * 256 + t];
    __syncthreads();
    if (t < HID) {
        float a = 0.f;
        for (int c = 0; c < 256; ++c) a += sp[c] * net_w[t * 256 + c];
        sh[t] = fmaxf(a, 0.f);
    }
    __syncthreads();
    if (t < NK) {
        float a = 0.f;
        for (int j = 0; j < HID; ++j) a += sh[j] * n_fc_w[t * HID + j];
        slog[t] = a * (1.0f / 30.0f);
    }
    __syncthreads();
    float m = slog[0];
    #pragma unroll
    for (int k = 1; k < NK; ++k) m = fmaxf(m, slog[k]);
    float e[NK], den = 0.f;
    #pragma unroll
    for (int k = 0; k < NK; ++k) { e[k] = __expf(slog[k] - m); den += e[k]; }
    float natt[NK];
    #pragma unroll
    for (int k = 0; k < NK; ++k) natt[k] = e[k] / den;
    if (t < NK) ws[OFF_NATT + b * NK + t] = natt[t];

    {
        float a1 = 0.f, a2 = 0.f;
        for (int j = 0; j < HID; ++j) {
            a1 += sh[j] * cin_fc_w[t * HID + j];
            a2 += sh[j] * out_fc_w[t * HID + j];
        }
        ws[OFF_CIN  + b * 256 + t] = 1.f / (1.f + __expf(-a1));
        ws[OFF_OUTA + b * 256 + t] = 1.f / (1.f + __expf(-a2));
    }
    if (t < 9) {
        float a = 0.f;
        for (int j = 0; j < HID; ++j) a += sh[j] * k2_fc_w[t * HID + j];
        ws[OFF_K2 + b * 9 + t] = 1.f / (1.f + __expf(-a));
    }
    {
        float a = natt[0] * base_b[t];
        #pragma unroll
        for (int k = 1; k < NK; ++k) a += natt[k] * extra_b[(k - 1) * 256 + t];
        ws[OFF_AGGB + b * 256 + t] = a;
    }
}

// ---------------------------------------------------------------------------
// 3) aggregated dynamic weights in bf16, layout [b][rs][o][c] (c contiguous)
// ---------------------------------------------------------------------------
__global__ __launch_bounds__(256) void k_weights(const float* __restrict__ base_w,
                                                 const float* __restrict__ ws,
                                                 unsigned short* __restrict__ wbf) {
    int idx = blockIdx.x * 256 + threadIdx.x;   // < 8*9*256*256
    int c   = idx & 255;
    int o   = (idx >> 8) & 255;
    int brs = idx >> 16;          // b*9 + rs
    int rs  = brs % 9;
    int b   = brs / 9;
    const float* bw = base_w + (size_t)(o * 256 + c) * 9;
    float acc = 0.f;
    #pragma unroll
    for (int k = 0; k < NK; ++k)
        acc += ws[OFF_NATT + b * NK + k] * bw[ROT[k][rs]];
    acc *= ws[OFF_CIN + b * 256 + c] * ws[OFF_K2 + b * 9 + rs] *
           ws[OFF_OUTA + b * 256 + o];
    wbf[idx] = f32_to_bf16_rne(acc);
}

// ---------------------------------------------------------------------------
// 4) implicit-GEMM conv, 9 shifted GEMMs, V_WMMA_F32_16X16X32_BF16
//    workgroup = (o-half 128, row h, sample b): M=128 x N=128 GEMM tile
//    8 waves as 4(M) x 2(N); each wave: 2 M-tiles x 4 N-tiles (8 accumulators)
//    Double-buffered LDS; staging of K-block kb+1 interleaved into the 9-tap
//    loop over kb. Staging is branch/division free: per-thread pixel geometry
//    precomputed once, OOB handled by clamped address x {0,1} mask,
//    non-existent second pixel stores to an LDS trash slot.
// ---------------------------------------------------------------------------
#define CHS    40               // LDS ushorts per pixel (32 ch + 8 pad) = 80 B
#define XPIX   (3 * 130)        // staged pixels per K-block (390)
#define TRASH  (XPIX * CHS)     // dummy LDS slot for inactive second pixel

__global__ __launch_bounds__(256, 2) void k_conv(const float* __restrict__ x,
                                                 const unsigned short* __restrict__ wbf,
                                                 const float* __restrict__ ws,
                                                 float* __restrict__ y) {
    __shared__ __align__(32) unsigned short xs[2][XPIX * CHS + CHS]; // 2 x 31,280 B
    __shared__ float sbias[128];

    const int ohalf = blockIdx.x;        // 0..1
    const int h     = blockIdx.y;
    const int b     = blockIdx.z;
    const int t     = threadIdx.x;
    const int obase = ohalf * 128;

    if (t < 128) sbias[t] = ws[OFF_AGGB + b * 256 + obase + t];

    const int lane  = t & 31;
    const int wave  = t >> 5;
    const int n     = lane & 15;    // fragment row (A) / column (B)
    const int kh    = lane >> 4;    // K-half selector
    const int wm    = wave & 3;     // M group: 32 output channels
    const int wn    = wave >> 2;    // N group: 64 pixels
    const int orow  = obase + wm * 32;
    const int pbase = wn * 64;

    const unsigned short* wb_b = wbf + (size_t)b * 9 * 65536;
    const float* xb = x + ((size_t)b * CIN) * (IMH * IMW);

    // ---- per-thread staging geometry, computed once -----------------------
    // pixel slot 0: p = t (always < 390); slot 1: p = t + 256 (only t < 134)
    const int p0   = t;
    const int row0 = p0 / 130;
    const int col0 = p0 - row0 * 130;
    const int hh0  = h + row0 - 1;
    const int ww0  = col0 - 1;
    const bool ok0 = ((unsigned)hh0 < (unsigned)IMH) && ((unsigned)ww0 < (unsigned)IMW);
    const float m0 = ok0 ? 1.f : 0.f;
    const int ga0  = (ok0 ? hh0 : 0) * IMW + (ok0 ? ww0 : 0);
    const int ls0  = p0 * CHS;

    const int p1   = t + 256;
    const bool ex1 = p1 < XPIX;
    const int row1 = p1 / 130;
    const int col1 = p1 - row1 * 130;
    const int hh1  = h + row1 - 1;
    const int ww1  = col1 - 1;
    const bool ok1 = ex1 && ((unsigned)hh1 < (unsigned)IMH) && ((unsigned)ww1 < (unsigned)IMW);
    const float m1 = ok1 ? 1.f : 0.f;
    const int ga1  = (ok1 ? hh1 : 0) * IMW + (ok1 ? ww1 : 0);
    const int ls1  = ex1 ? p1 * CHS : TRASH;

    v8f acc[2][4] = {};

    // ---- full stage of K-block 0 into buffer 0 ----------------------------
    {
        unsigned short* dst = xs[0];
        #pragma unroll
        for (int c = 0; c < 32; ++c) {
            float v0 = xb[(size_t)c * (IMH * IMW) + ga0] * m0;
            float v1 = xb[(size_t)c * (IMH * IMW) + ga1] * m1;
            dst[ls0 + c] = f32_to_bf16_rne(v0);
            dst[ls1 + c] = f32_to_bf16_rne(v1);
        }
    }
    __syncthreads();

    for (int kb = 0; kb < 8; ++kb) {
        const unsigned short* cur = xs[kb & 1];
        unsigned short* nxt = xs[(kb + 1) & 1];
        const int c0 = kb * 32;
        const bool more = (kb < 7);
        const unsigned short* wkb = wb_b + c0 + kh * 8;
        const float* xk = xb + (size_t)(c0 + 32) * (IMH * IMW); // next K-block

        #pragma unroll
        for (int rs = 0; rs < 9; ++rs) {
            constexpr int CB[10] = {0, 4, 8, 12, 16, 20, 23, 26, 29, 32};
            // ---- issue staging loads for this chunk of the next K-block
            float tv0[4], tv1[4];
            if (more) {
                #pragma unroll
                for (int q = 0; q < 4; ++q) {
                    const int c = CB[rs] + q;
                    if (c < CB[rs + 1]) {
                        tv0[q] = xk[(size_t)c * (IMH * IMW) + ga0] * m0;
                        tv1[q] = xk[(size_t)c * (IMH * IMW) + ga1] * m1;
                    }
                }
            }

            // ---- A fragments for this wave's two M-tiles ([b][rs][o][c])
            const int r = rs / 3, s = rs - r * 3;
            const unsigned short* a0p =
                wkb + (size_t)rs * 65536 + (size_t)(orow + n) * 256;
            const unsigned short* a1p = a0p + 16 * 256;
            v8bf a0l = *(const v8bf*)(a0p);
            v8bf a0h = *(const v8bf*)(a0p + 16);
            v8bf a1l = *(const v8bf*)(a1p);
            v8bf a1h = *(const v8bf*)(a1p + 16);
            v16bf A0 = SHUF16(a0l, a0h);
            v16bf A1 = SHUF16(a1l, a1h);

            // ---- preload all 4 B fragments, then 8 WMMAs
            const int prow = r * 130 + s;
            v16bf Bf[4];
            #pragma unroll
            for (int nt = 0; nt < 4; ++nt) {
                const unsigned short* bp =
                    &cur[(prow + pbase + nt * 16 + n) * CHS + kh * 8];
                v8bf bl = *(const v8bf*)(bp);
                v8bf bh = *(const v8bf*)(bp + 16);
                Bf[nt] = SHUF16(bl, bh);
            }
            #pragma unroll
            for (int nt = 0; nt < 4; ++nt) {
                acc[0][nt] = __builtin_amdgcn_wmma_f32_16x16x32_bf16(
                    false, A0, false, Bf[nt], (short)0, acc[0][nt], false, false);
                acc[1][nt] = __builtin_amdgcn_wmma_f32_16x16x32_bf16(
                    false, A1, false, Bf[nt], (short)0, acc[1][nt], false, false);
            }

            // ---- commit staging chunk (unconditional stores, no branches)
            if (more) {
                #pragma unroll
                for (int q = 0; q < 4; ++q) {
                    const int c = CB[rs] + q;
                    if (c < CB[rs + 1]) {
                        nxt[ls0 + c] = f32_to_bf16_rne(tv0[q]);
                        nxt[ls1 + c] = f32_to_bf16_rne(tv1[q]);
                    }
                }
            }
        }
        __syncthreads();   // next buffer complete; current buffer reusable
    }

    // ---- epilogue: C/D element v holds row M = v + 8*kh, column N = n
    float* yrow = y + (((size_t)b * COUT) * IMH + h) * IMW;
    #pragma unroll
    for (int mt = 0; mt < 2; ++mt) {
        #pragma unroll
        for (int nt = 0; nt < 4; ++nt) {
            const int px = pbase + nt * 16 + n;
            #pragma unroll
            for (int v = 0; v < 8; ++v) {
                const int o = orow + mt * 16 + v + 8 * kh;
                yrow[(size_t)o * (IMH * IMW) + px] = acc[mt][nt][v] + sbias[o - obase];
            }
        }
    }
}

// ---------------------------------------------------------------------------
extern "C" void kernel_launch(void* const* d_in, const int* in_sizes, int n_in,
                              void* d_out, int out_size, void* d_ws, size_t ws_size,
                              hipStream_t stream) {
    const float* x        = (const float*)d_in[0];
    const float* net_w    = (const float*)d_in[1];
    const float* n_fc_w   = (const float*)d_in[2];
    const float* cin_fc_w = (const float*)d_in[3];
    const float* k2_fc_w  = (const float*)d_in[4];
    const float* out_fc_w = (const float*)d_in[5];
    const float* base_w   = (const float*)d_in[6];
    const float* base_b   = (const float*)d_in[7];
    const float* extra_b  = (const float*)d_in[8];

    float* wsf = (float*)d_ws;
    unsigned short* wbf = (unsigned short*)((char*)d_ws + OFF_WBF_BYTES);
    float* out = (float*)d_out;

    k_pool<<<BS * CIN, 256, 0, stream>>>(x, wsf);
    k_attn<<<BS, 256, 0, stream>>>(net_w, n_fc_w, cin_fc_w, k2_fc_w, out_fc_w,
                                   base_b, extra_b, wsf);
    k_weights<<<(BS * 9 * COUT * CIN) / 256, 256, 0, stream>>>(base_w, wsf, wbf);
    dim3 grid(2, IMH, BS);
    k_conv<<<grid, 256, 0, stream>>>(x, wbf, wsf, out);
}
